// LPKT4LPR_15152644620310
// MI455X (gfx1250) — matrix operand order repaired and checked by reading, not compiled, not run on hardware
//
#include <hip/hip_runtime.h>
#include <hip/hip_bf16.h>

#define B_  64
#define S_  128
#define NQ_ 2000
#define C_  128
#define K_  64
#define DE_ 64

#define CP_ 132  // fp32 h column-count stride for transposed state hsh[k][c]
                 // (132%64==4 -> conflict-free strided column reads; 132%4==0 -> 16B-aligned rows)
#define HB_ 72   // bf16 h row stride (144B: 16B-aligned chunks, conflict-free frag reads)

typedef __bf16 v16bf __attribute__((ext_vector_type(16)));
typedef __bf16 v8bf  __attribute__((ext_vector_type(8)));
typedef float  v8f   __attribute__((ext_vector_type(8)));
typedef float  v4f   __attribute__((ext_vector_type(4)));

#define CAT16(lo, hi) __builtin_shufflevector(lo, hi, 0,1,2,3,4,5,6,7,8,9,10,11,12,13,14,15)

// Fast transcendentals: gfx1250 has a hardware v_tanh_f32 (TRANS op). Prefer it;
// otherwise fall back to raw v_rcp_f32 (no IEEE div expansion).
__device__ __forceinline__ float fast_tanh(float x) {
#if __has_builtin(__builtin_amdgcn_tanhf)
  return __builtin_amdgcn_tanhf(x);
#else
  float e = __expf(2.0f * x);                       // tanh(x) = 1 - 2/(e^{2x}+1)
  return 1.0f - 2.0f * __builtin_amdgcn_rcpf(e + 1.0f);
#endif
}

__device__ __forceinline__ float sigmoidf_(float x) {
#if __has_builtin(__builtin_amdgcn_tanhf)
  return __builtin_fmaf(0.5f, __builtin_amdgcn_tanhf(0.5f * x), 0.5f);
#else
  return __builtin_amdgcn_rcpf(1.0f + __expf(-x));
#endif
}

// Round-to-nearest-even conversion: used for weights / one-time init.
__device__ __forceinline__ __bf16 f2bf(float f) {
  union { float f; unsigned u; } v; v.f = f;
  unsigned r = v.u + 0x7FFFu + ((v.u >> 16) & 1u);
  union { unsigned short s; __bf16 b; } o; o.s = (unsigned short)(r >> 16);
  return o.b;
}

// Truncating conversion: inner-loop activations. Lowers to a plain high-half
// store (ds_store_b16_d16_hi) with zero ALU work.
__device__ __forceinline__ __bf16 f2bf_fast(float f) {
  union { float f; unsigned u; } v; v.f = f;
  union { unsigned short s; __bf16 b; } o; o.s = (unsigned short)(v.u >> 16);
  return o.b;
}

// ---------------- prep kernels (hoist question-indexed work out of the scan) --------------

// all_learning[b,s,k] = concat(E_q[q], E_c[corr]) @ W1 + b1
__global__ __launch_bounds__(64)
void lpkt_prep_learning(const int* __restrict__ qseq, const int* __restrict__ cseq,
                        const float* __restrict__ Eq, const float* __restrict__ Ec,
                        const float* __restrict__ W1, const float* __restrict__ b1,
                        float* __restrict__ AL) {
  const int bs = blockIdx.x;          // b*S + s
  const int k  = threadIdx.x;         // 0..63
  const int q  = qseq[bs];
  const int cr = cseq[bs];
  float acc = b1[k];
  #pragma unroll 4
  for (int d = 0; d < DE_; ++d) acc += Eq[q * DE_ + d] * W1[d * K_ + k];
  #pragma unroll 4
  for (int d = 0; d < 64; ++d)  acc += Ec[cr * 64 + d] * W1[(DE_ + d) * K_ + k];
  AL[bs * K_ + k] = acc;
}

// Qdiff[q,c] = sigmoid(E_q[q]@Wdiff + bdiff),  Qdisc[q] = sigmoid(E_q[q]@Wdisc + bdisc)*5
__global__ __launch_bounds__(128)
void lpkt_prep_qdiff(const float* __restrict__ Eq, const float* __restrict__ Wdiff,
                     const float* __restrict__ bdiff, const float* __restrict__ Wdisc,
                     const float* __restrict__ bdisc,
                     float* __restrict__ Qdiff, float* __restrict__ Qdisc) {
  const int q = blockIdx.x;
  const int c = threadIdx.x;
  float acc = bdiff[c];
  #pragma unroll 4
  for (int d = 0; d < DE_; ++d) acc += Eq[q * DE_ + d] * Wdiff[d * C_ + c];
  Qdiff[q * C_ + c] = sigmoidf_(acc);
  if (c == 0) {
    float a = bdisc[0];
    for (int d = 0; d < DE_; ++d) a += Eq[q * DE_ + d] * Wdisc[d];
    Qdisc[q] = sigmoidf_(a) * 5.0f;
  }
}

// W4T[n,k] = bf16(W4[k,n]) for k<64 (the h_pre half of W4), row-major 64x64
__global__ __launch_bounds__(256)
void lpkt_prep_w4t(const float* __restrict__ W4, __bf16* __restrict__ W4T) {
  int i = blockIdx.x * blockDim.x + threadIdx.x;
  if (i < 64 * 64) {
    int n = i >> 6, k = i & 63;
    W4T[n * 64 + k] = f2bf(W4[k * 64 + n]);
  }
}

// ---------------- main persistent scan: one workgroup per batch element ----------------
// 256 threads = 8 waves; wave w owns h rows (c) [16w, 16w+16). Per step:
//   A : stage q_e / q_next / diff rows + cat;  gate matvecs (W2|W3 staged in LDS);  LGW4 vector
//   B : WMMA  h_pre(bf16)[128x64] @ W4topT(bf16)[64x64] -> gamma_f, fused vectorized h update
//   C : ability dot + h_tilde reduction;  D: scalar y -> pred
// State: hsh = fp32 h TRANSPOSED [k][CP_] (vector LDS I/O in the update),
//        hbf = bf16 h row-major [c][HB_] (WMMA A operand fragments).
__global__ __launch_bounds__(256)
void lpkt_scan(const int* __restrict__ qseq, const float* __restrict__ qmat,
               const float* __restrict__ h0g,
               const float* __restrict__ W2, const float* __restrict__ b2g,
               const float* __restrict__ W3, const float* __restrict__ b3g,
               const float* __restrict__ W4, const float* __restrict__ b4g,
               const float* __restrict__ Wab, const float* __restrict__ bab,
               const float* __restrict__ AL, const float* __restrict__ Qdiff,
               const float* __restrict__ Qdisc, const __bf16* __restrict__ W4T,
               float* __restrict__ pred) {
  extern __shared__ __align__(16) char smem_raw[];
  float* W23     = (float*)smem_raw;        // [192][128]  cols 0-63 = W2, 64-127 = W3
  float* hsh     = W23 + 192 * 128;         // [K_][CP_]   fp32 state h, TRANSPOSED
  float* W4loL   = hsh + K_ * CP_;          // [64][64]    W4 rows 64..127 (LG half)
  float* WabL    = W4loL + 64 * 64;         // [64]
  float* b2L     = WabL + 64;               // [64]
  float* b3L     = b2L + 64;                // [64]
  float* b4L     = b3L + 64;                // [64]
  float* catL    = b4L + 64;                // [192]
  float* LGl     = catL + 192;              // [64]
  float* LGW4    = LGl + 64;                // [64]
  float* hTld    = LGW4 + 64;               // [64]
  float* qeL     = hTld + 64;               // [128]
  float* qnL     = qeL + 128;               // [128]
  float* dfL     = qnL + 128;               // [128]
  float* contrib = dfL + 128;               // [128]
  __bf16* hbf    = (__bf16*)(contrib + 128); // [C_][HB_] bf16 mirror of h (WMMA A operand)

  const int b    = blockIdx.x;
  const int tid  = threadIdx.x;
  const int wv   = tid >> 5;
  const int lane = tid & 31;
  const int hf   = lane >> 4;    // lane half (A/B fragment K selector)
  const int ln   = lane & 15;

  // ---- one-time staging ----
  for (int i = tid; i < 192 * 64; i += 256) {
    int j = i >> 6, k = i & 63;
    W23[j * 128 + k]      = W2[i];
    W23[j * 128 + 64 + k] = W3[i];
  }
  for (int i = tid; i < 64 * 64; i += 256) W4loL[i] = W4[64 * 64 + i];
  if (tid < 64) { WabL[tid] = Wab[tid]; b2L[tid] = b2g[tid]; b3L[tid] = b3g[tid]; b4L[tid] = b4g[tid]; }
  for (int i = tid; i < C_ * K_; i += 256) {
    int c = i >> 6, k = i & 63;
    float v = h0g[i];
    hsh[k * CP_ + c] = v;        // transposed store
    hbf[c * HB_ + k] = f2bf(v);
  }
  {
    int q0 = qseq[b * S_];
    if (tid < 128) qnL[tid] = qmat[q0 * C_ + tid];
    if (tid == 0)  pred[b * S_] = 0.0f;
  }
  __syncthreads();
  // h_tilde0 = q_seq[:,0] @ h0
  if (tid < 64) {
    float acc = 0.0f;
    const float* hrow = &hsh[tid * CP_];
    for (int c = 0; c < C_; c += 4) {
      v4f hv = *(const v4f*)&hrow[c];
      v4f qv = *(const v4f*)&qnL[c];
      acc += hv[0]*qv[0] + hv[1]*qv[1] + hv[2]*qv[2] + hv[3]*qv[3];
    }
    hTld[tid] = acc;
  }

  // ---- preload B fragments of W4topT (constant across all 127 steps) ----
  // B frags in this code's contiguous k order (same as A): lane col = ln, half0 holds K=kt*32+0..15, half1 K=kt*32+16..31.
  v16bf Bf[4][2];
  #pragma unroll
  for (int nt = 0; nt < 4; ++nt)
    #pragma unroll
    for (int kt = 0; kt < 2; ++kt) {
      const __bf16* p = W4T + (nt * 16 + ln) * 64 + kt * 32 + hf * 16;
      v8bf lo = *(const v8bf*)p;
      v8bf hi = *(const v8bf*)(p + 8);
      Bf[nt][kt] = CAT16(lo, hi);
    }
  __syncthreads();

  const float babv = bab[0];

  for (int s = 0; s < S_ - 1; ++s) {
    const int qe_id = qseq[b * S_ + s];
    const int qn_id = qseq[b * S_ + s + 1];

    // ---- phase A: stage per-step rows + cat ----
    if (tid < 128) {
      qeL[tid] = qmat[qe_id * C_ + tid];
      dfL[tid] = Qdiff[qn_id * C_ + tid];
    } else {
      qnL[tid - 128] = qmat[qn_id * C_ + (tid - 128)];
    }
    if (tid < 64)       catL[tid] = (s == 0) ? 0.0f : AL[(b * S_ + s - 1) * K_ + tid];
    else if (tid < 128) catL[tid] = AL[(b * S_ + s) * K_ + (tid - 64)];
    else if (tid < 192) catL[tid] = hTld[tid - 128];
    if (tid == 0 && s + 2 < S_) {   // warm next step's rows into cache
      int qp = qseq[b * S_ + s + 2];
      __builtin_prefetch(qmat + (size_t)qp * C_, 0, 3);
      __builtin_prefetch(Qdiff + (size_t)qp * C_, 0, 3);
    }
    __syncthreads();

    // ---- phase A1: learning_gain / gamma_l -> LG ----
    if (tid < 64) {
      float a2 = b2L[tid], a3 = b3L[tid];
      #pragma unroll 4
      for (int j = 0; j < 192; ++j) {
        float cv = catL[j];
        a2 += cv * W23[j * 128 + tid];
        a3 += cv * W23[j * 128 + 64 + tid];
      }
      float gain = fast_tanh(a2);
      float gl   = sigmoidf_(a3);
      LGl[tid]   = gl * (gain + 1.0f) * 0.5f;
    }
    __syncthreads();

    // ---- phase A2: LGW4 = LG @ W4[64:128] + b4 (shared across all C rows) ----
    if (tid < 64) {
      float acc = b4L[tid];
      #pragma unroll 4
      for (int k = 0; k < 64; ++k) acc += LGl[k] * W4loL[k * 64 + tid];
      LGW4[tid] = acc;
    }
    __syncthreads();

    // ---- phase B: WMMA  gamma_f = sigmoid(h_pre @ W4top + LGW4);  fused vector h update ----
    {
      const int arow = 16 * wv + ln;                 // A frag: lanes 0-15 & 16-31 both hold row M=ln
      const __bf16* abase = &hbf[arow * HB_];
      v8bf l0 = *(const v8bf*)(abase + hf * 16);     // K = hf*16 .. +7
      v8bf u0 = *(const v8bf*)(abase + hf * 16 + 8); // K = hf*16 + 8 .. +7
      v16bf a0 = CAT16(l0, u0);
      v8bf l1 = *(const v8bf*)(abase + 32 + hf * 16);
      v8bf u1 = *(const v8bf*)(abase + 40 + hf * 16);
      v16bf a1 = CAT16(l1, u1);

      const int c0 = 16 * wv + 8 * hf;               // this thread's 8 contiguous c rows
      v4f qe0 = *(const v4f*)&qeL[c0];
      v4f qe1 = *(const v4f*)&qeL[c0 + 4];

      #pragma unroll
      for (int nt = 0; nt < 4; ++nt) {
        v8f acc = {};
        acc = __builtin_amdgcn_wmma_f32_16x16x32_bf16(false, a0, false, Bf[nt][0],
                                                      (short)0, acc, false, false);
        acc = __builtin_amdgcn_wmma_f32_16x16x32_bf16(false, a1, false, Bf[nt][1],
                                                      (short)0, acc, false, false);
        const int n     = nt * 16 + ln;
        const float lgw = LGW4[n];
        const float lg  = LGl[n];

        float* hrow = &hsh[n * CP_ + c0];            // 8 contiguous fp32 (16B aligned)
        v4f hold0 = *(const v4f*)hrow;
        v4f hold1 = *(const v4f*)(hrow + 4);
        v4f hn0, hn1;
        #pragma unroll
        for (int r = 0; r < 4; ++r) {
          float gf = sigmoidf_(acc[r] + lgw);        // acc[r] <-> c0+r (both lane halves)
          hn0[r] = qe0[r] * lg + gf * hold0[r];
        }
        #pragma unroll
        for (int r = 0; r < 4; ++r) {
          float gf = sigmoidf_(acc[4 + r] + lgw);
          hn1[r] = qe1[r] * lg + gf * hold1[r];
        }
        *(v4f*)hrow       = hn0;
        *(v4f*)(hrow + 4) = hn1;
        #pragma unroll
        for (int r = 0; r < 4; ++r) hbf[(c0 + r) * HB_ + n]     = f2bf_fast(hn0[r]);
        #pragma unroll
        for (int r = 0; r < 4; ++r) hbf[(c0 + 4 + r) * HB_ + n] = f2bf_fast(hn1[r]);
      }
    }
    __syncthreads();

    // ---- phase C: ability/diff contributions + h_tilde reduction ----
    if (tid < 128) {
      const int c = tid;
      const float qn = qnL[c];
      float dot = babv;
      #pragma unroll 8
      for (int k = 0; k < 64; ++k) dot += hsh[k * CP_ + c] * WabL[k];
      contrib[c] = (sigmoidf_(dot) - dfL[c]) * qn;
    } else if (tid < 192) {
      const int k = tid - 128;
      float acc = 0.0f;
      const float* hrow = &hsh[k * CP_];
      for (int c = 0; c < C_; c += 4) {
        v4f hv = *(const v4f*)&hrow[c];
        v4f qv = *(const v4f*)&qnL[c];
        acc += hv[0]*qv[0] + hv[1]*qv[1] + hv[2]*qv[2] + hv[3]*qv[3];
      }
      hTld[k] = acc;
    }
    __syncthreads();

    // ---- phase D: y = sigmoid(disc * sum_c contrib) ----
    if (tid == 0) {
      float sum = 0.0f;
      for (int c = 0; c < C_; ++c) sum += contrib[c];
      pred[b * S_ + s + 1] = sigmoidf_(Qdisc[qn_id] * sum);
    }
    __syncthreads();
  }
}

// ---------------- host-side launcher ----------------
extern "C" void kernel_launch(void* const* d_in, const int* in_sizes, int n_in,
                              void* d_out, int out_size, void* d_ws, size_t ws_size,
                              hipStream_t stream) {
  const int*   qseq  = (const int*)d_in[0];
  const int*   cseq  = (const int*)d_in[1];
  const float* qmat  = (const float*)d_in[2];
  const float* Eq    = (const float*)d_in[3];
  const float* Ec    = (const float*)d_in[4];
  const float* h0    = (const float*)d_in[5];
  const float* W1    = (const float*)d_in[6];
  const float* b1    = (const float*)d_in[7];
  const float* W2    = (const float*)d_in[8];
  const float* b2    = (const float*)d_in[9];
  const float* W3    = (const float*)d_in[10];
  const float* b3    = (const float*)d_in[11];
  const float* W4    = (const float*)d_in[12];
  const float* b4    = (const float*)d_in[13];
  const float* Wab   = (const float*)d_in[14];
  const float* bab   = (const float*)d_in[15];
  const float* Wdiff = (const float*)d_in[16];
  const float* bdiff = (const float*)d_in[17];
  const float* Wdisc = (const float*)d_in[18];
  const float* bdisc = (const float*)d_in[19];
  float* pred = (float*)d_out;

  float* ws   = (float*)d_ws;
  float* AL   = ws;                                   // B*S*K  = 524288 f32
  float* Qd   = AL + (size_t)B_ * S_ * K_;            // NQ*C   = 256000 f32
  float* Qs   = Qd + (size_t)NQ_ * C_;                // NQ     = 2000 f32 (pad to 2048)
  __bf16* W4T = (__bf16*)(Qs + 2048);                 // 64*64 bf16 (16B-aligned)

  lpkt_prep_learning<<<dim3(B_ * S_), dim3(64), 0, stream>>>(qseq, cseq, Eq, Ec, W1, b1, AL);
  lpkt_prep_qdiff<<<dim3(NQ_), dim3(128), 0, stream>>>(Eq, Wdiff, bdiff, Wdisc, bdisc, Qd, Qs);
  lpkt_prep_w4t<<<dim3(16), dim3(256), 0, stream>>>(W4, W4T);

  // dynamic LDS:
  // (192*128 + 64*132 + 64*64 + 64*4 + 192 + 64*3 + 128*4) * 4B  +  128*72 * 2B
  const size_t smem_bytes =
      (size_t)(192 * 128 + K_ * CP_ + 64 * 64 + 64 * 4 + 192 + 64 * 3 + 128 * 4) * sizeof(float)
      + (size_t)C_ * HB_ * sizeof(__bf16);
  lpkt_scan<<<dim3(B_), dim3(256), smem_bytes, stream>>>(
      qseq, qmat, h0, W2, b2, W3, b3, W4, b4, Wab, bab, AL, Qd, Qs, W4T, pred);
}
